// FocalRegionLoss_67869073211922
// MI455X (gfx1250) — matrix operand order repaired and checked by reading, not compiled, not run on hardware
//
#include <hip/hip_runtime.h>

// ---------------------------------------------------------------------------
// FocalRegionLoss on MI455X (gfx1250, wave32)
//
// Roofline: 117 MB streamed @ 23.3 TB/s => ~5 us floor; ~10 flops/pixel =>
// purely memory-bound, so: one streaming pass (pass1) + tiny finalize (pass2).
//
// pass1: per-(batch,class) segment sums of channel-summed |in-tgt| and exact
//   pixel counts via one-hot WMMA (v_wmma_f32_16x16x32_f16):
//     D_j[m][n] = sum_k onehot(cls_k == 16j+m) * (n<8 ? csum_k : 1.0)
//   -> columns 0..7 = value sums, 8..15 = exact counts, f32 accumulation.
//   Mask tile staged global->LDS with GLOBAL_LOAD_ASYNC_TO_LDS_B128 (ASYNCcnt);
//   in/tgt streamed as b128 loads; next tile prefetched (global_prefetch_b8).
//
// pass2: weight is constant per segment, so the loss collapses to
//   sum_seg f(seg)*sums(seg) / (B*C*H*W) -- no second pass over the 100MB.
//
// Occupancy: 512 blocks (2/WGP, 4 waves/SIMD) so in-flight loads comfortably
// exceed the ~11.6 MB bandwidth-delay product of 23.3 TB/s HBM.
// Deterministic: no atomics; per-block partials in d_ws, fixed-order sums.
// ---------------------------------------------------------------------------

typedef __attribute__((ext_vector_type(16))) _Float16 v16h;
typedef __attribute__((ext_vector_type(8)))  float    v8f;
typedef int v4i __attribute__((vector_size(16)));   // matches builtin param type

#define BB   16          // batch
#define CC   3           // channels
#define HW   262144      // 512*512
#define NC   64          // classes per image
#define CHUNKS_PER_IMG 32
#define PIX_PER_BLOCK  8192    // HW / CHUNKS_PER_IMG
#define TILE_PIX       2048
#define NTILES         4       // PIX_PER_BLOCK / TILE_PIX
#define NTOT           12582912.0f  // B*C*H*W

#if __has_builtin(__builtin_amdgcn_global_load_async_to_lds_b128) && \
    __has_builtin(__builtin_amdgcn_s_wait_asynccnt)
#define HAVE_ASYNC 1
#else
#define HAVE_ASYNC 0
#endif

#define AS1 __attribute__((address_space(1)))
#define AS3 __attribute__((address_space(3)))

// flat->LDS offset: low 32 bits of the flat address are the LDS byte offset
// (ISA 10.2 aperture mapping); flat global == as1 numerically.
static __device__ __forceinline__ AS3 v4i* lds_cast4(void* p) {
  return (AS3 v4i*)(unsigned int)(unsigned long long)p;
}
static __device__ __forceinline__ AS1 v4i* gbl_cast4(const void* p) {
  return (AS1 v4i*)(unsigned long long)p;
}

// ws layout: 512 blocks * 128 floats  (slots 0..63 = value sums, 64..127 = counts)
__global__ __launch_bounds__(256)
void fr_pass1(const float* __restrict__ inp, const float* __restrict__ tgt,
              const int* __restrict__ mask, float* __restrict__ ws)
{
  __shared__ __align__(16) float sCsum[TILE_PIX];  // channel-summed |in-tgt|
  __shared__ __align__(16) int   sCls [TILE_PIX];  // class id (raw mask, 0..63)

  const int tid   = threadIdx.x;
  const int lane  = tid & 31;
  const int wid   = tid >> 5;           // 8 waves per block
  const int m     = lane & 15;          // A row / B column index
  const int khalf = lane >> 4;          // which K half this lane holds

  const int bimg  = blockIdx.y;                       // image
  const int base  = blockIdx.x * PIX_PER_BLOCK;       // pixel base in image
  const float* ip0 = inp + (size_t)bimg * CC * HW;
  const float* tp0 = tgt + (size_t)bimg * CC * HW;
  const int*   mp0 = mask + (size_t)bimg * HW;

  // per-wave WMMA accumulators: 4 x (16x16 f32) covering classes j*16+m
  v8f dacc[4];
#pragma unroll
  for (int j = 0; j < 4; ++j)
#pragma unroll
    for (int r = 0; r < 8; ++r) dacc[j][r] = 0.0f;

  for (int t = 0; t < NTILES; ++t) {
    const int tBase = base + t * TILE_PIX;
    const int l0 = tid * 8;            // 8 consecutive pixels per thread
    const int p0 = tBase + l0;

    // ---- Phase A: stage tile into LDS ----------------------------------
#if HAVE_ASYNC
    // mask: async DMA global->LDS, bypasses VGPRs, overlaps with csum math
    __builtin_amdgcn_global_load_async_to_lds_b128(
        gbl_cast4(mp0 + p0),     lds_cast4(&sCls[l0]),     0, 0);
    __builtin_amdgcn_global_load_async_to_lds_b128(
        gbl_cast4(mp0 + p0 + 4), lds_cast4(&sCls[l0 + 4]), 0, 0);
#else
    *(int4*)&sCls[l0]     = *(const int4*)(mp0 + p0);
    *(int4*)&sCls[l0 + 4] = *(const int4*)(mp0 + p0 + 4);
#endif

#pragma unroll
    for (int q = 0; q < 2; ++q) {
      const int l = l0 + q * 4;
      const int p = tBase + l;
      float4 xa = *(const float4*)(ip0 + p);
      float4 xb = *(const float4*)(tp0 + p);
      float4 ya = *(const float4*)(ip0 + p + HW);
      float4 yb = *(const float4*)(tp0 + p + HW);
      float4 za = *(const float4*)(ip0 + p + 2 * HW);
      float4 zb = *(const float4*)(tp0 + p + 2 * HW);
      float4 s;
      s.x = fabsf(xa.x - xb.x) + fabsf(ya.x - yb.x) + fabsf(za.x - zb.x);
      s.y = fabsf(xa.y - xb.y) + fabsf(ya.y - yb.y) + fabsf(za.y - zb.y);
      s.z = fabsf(xa.z - xb.z) + fabsf(ya.z - yb.z) + fabsf(za.z - zb.z);
      s.w = fabsf(xa.w - xb.w) + fabsf(ya.w - yb.w) + fabsf(za.w - zb.w);
      *(float4*)&sCsum[l] = s;
    }

    if (t + 1 < NTILES) {   // global_prefetch_b8 for next tile
      __builtin_prefetch(ip0 + p0 + TILE_PIX, 0, 1);
      __builtin_prefetch(tp0 + p0 + TILE_PIX, 0, 1);
      __builtin_prefetch(mp0 + p0 + TILE_PIX, 0, 1);
    }

#if HAVE_ASYNC
    __builtin_amdgcn_s_wait_asynccnt(0);
#endif
    __syncthreads();

    // ---- Phase B: one-hot WMMA binning, 8 groups of 32 pixels per wave --
    const bool valCol = (m < 8);  // B cols 0..7 carry values, 8..15 carry 1.0
#pragma unroll 1
    for (int g = 0; g < 8; ++g) {
      const int off = wid * 256 + g * 32;   // 32 pixels of this group

      // B (32x16 f16): lane holds column m, K = h + 16*khalf  (same order as A)
      v16h bm;
#pragma unroll
      for (int h = 0; h < 16; ++h) {
        float vv = sCsum[off + (khalf << 4) + h];
        bm[h] = valCol ? (_Float16)vv : (_Float16)1.0f;
      }

      // A_j (16x32 f16): lane holds row m; K(h) = h + 16*khalf
      v16h a[4];
#pragma unroll
      for (int h = 0; h < 16; ++h) {
        const int k  = (khalf << 4) + h;
        const int c  = sCls[off + k];        // guaranteed 0..63
        const int hi = c >> 4;
        const _Float16 on = ((c & 15) == m) ? (_Float16)1.0f : (_Float16)0.0f;
        a[0][h] = (hi == 0) ? on : (_Float16)0.0f;
        a[1][h] = (hi == 1) ? on : (_Float16)0.0f;
        a[2][h] = (hi == 2) ? on : (_Float16)0.0f;
        a[3][h] = (hi == 3) ? on : (_Float16)0.0f;
      }

#pragma unroll
      for (int j = 0; j < 4; ++j) {
        dacc[j] = __builtin_amdgcn_wmma_f32_16x16x32_f16(
            /*neg_a=*/false, a[j], /*neg_b=*/false, bm,
            /*c_mod=*/(short)0, dacc[j], /*reuse_a=*/false, /*reuse_b=*/false);
      }
    }
    __syncthreads();
  }

  // ---- Extract per-wave results into LDS, deterministic block reduce ----
  // D layout (32-bit 16x16): VGPR r: lanes 0-15 -> (M=r, N=lane),
  //                                  lanes 16-31 -> (M=r+8, N=lane-16).
  float* part = sCsum;  // reuse: 8 waves * 128 slots = 1024 floats
  int slotBase = -1;
  if      (lane == 0)  slotBase = 0;    // value sums, cls j*16 + r
  else if (lane == 16) slotBase = 8;    // value sums, cls j*16 + 8 + r
  else if (lane == 8)  slotBase = 64;   // counts,     cls j*16 + r
  else if (lane == 24) slotBase = 72;   // counts,     cls j*16 + 8 + r
  if (slotBase >= 0) {
#pragma unroll
    for (int j = 0; j < 4; ++j)
#pragma unroll
      for (int r = 0; r < 8; ++r)
        part[wid * 128 + slotBase + j * 16 + r] = dacc[j][r];
  }
  __syncthreads();

  if (tid < 128) {
    float acc = 0.0f;
#pragma unroll
    for (int w = 0; w < 8; ++w) acc += part[w * 128 + tid];
    const int blockId = blockIdx.y * CHUNKS_PER_IMG + blockIdx.x;
    ws[blockId * 128 + tid] = acc;
  }
}

// ---- finalize: 1024 segments -> scalar --------------------------------
__global__ __launch_bounds__(1024)
void fr_pass2(const float* __restrict__ ws, float* __restrict__ out)
{
  __shared__ float red[1024];
  const int tid  = threadIdx.x;          // segment id = bimg*64 + cls
  const int bimg = tid >> 6;
  const int cls  = tid & 63;

  float s = 0.0f, cn = 0.0f;
#pragma unroll
  for (int bx = 0; bx < CHUNKS_PER_IMG; ++bx) {
    const float* p = ws + (size_t)(bimg * CHUNKS_PER_IMG + bx) * 128;
    s  += p[cls];
    cn += p[64 + cls];
  }
  const float avg = s / fmaxf(cn * (float)CC, 1.0f);

  red[tid] = avg;
  __syncthreads();
  for (int st = 512; st > 0; st >>= 1) {
    if (tid < st) red[tid] = fmaxf(red[tid], red[tid + st]);
    __syncthreads();
  }
  float wmax = red[0];
  __syncthreads();
  if (!(wmax > 0.0f)) wmax = 1.0f;

  const float f = fminf(fmaxf(avg / wmax, 0.0f), 1.0f) * 1.0f /*BETA*/ + 1.0f;
  red[tid] = s * f;
  __syncthreads();
  for (int st = 512; st > 0; st >>= 1) {
    if (tid < st) red[tid] += red[tid + st];
    __syncthreads();
  }
  if (tid == 0) out[0] = red[0] / NTOT;
}

extern "C" void kernel_launch(void* const* d_in, const int* in_sizes, int n_in,
                              void* d_out, int out_size, void* d_ws, size_t ws_size,
                              hipStream_t stream)
{
  const float* inp  = (const float*)d_in[0];
  const float* tgt  = (const float*)d_in[1];
  const int*   mask = (const int*)d_in[2];
  float* out = (float*)d_out;
  float* ws  = (float*)d_ws;   // 512 blocks * 128 floats = 256 KB

  dim3 g1(CHUNKS_PER_IMG, BB);
  fr_pass1<<<g1, 256, 0, stream>>>(inp, tgt, mask, ws);
  fr_pass2<<<1, 1024, 0, stream>>>(ws, out);
}
